// HierarchicalWTABlockV4_27144193310740
// MI455X (gfx1250) — matrix operand order repaired and checked by reading, not compiled, not run on hardware
//
#include <hip/hip_runtime.h>
#include <hip/hip_bf16.h>
#include <math.h>

// ---------------------------------------------------------------------------
// HierarchicalWTABlock for MI455X (gfx1250, wave32, WMMA bf16 16x16x32)
// 32-token M-tiles: each B-fragment load feeds 2 WMMAs, halves L2 traffic.
// ---------------------------------------------------------------------------

typedef __attribute__((ext_vector_type(16))) __bf16 v16bf;
typedef __attribute__((ext_vector_type(8)))  __bf16 v8bf;
typedef __attribute__((ext_vector_type(8)))  float  v8f;

#define B_      4
#define L_      4096
#define D_      512
#define TWO_D   1024
#define G_      32
#define K_      32
#define N_SLOTS 1024
#define LN_EPS  1e-5f
#define TM      32              // rows (tokens/slots) per block

// -------------------------- device helpers ---------------------------------

__device__ __forceinline__ float gelu_erf(float x) {
    return 0.5f * x * (1.0f + erff(x * 0.70710678118654752f));
}

__device__ __forceinline__ v8f zero8() {
    v8f z;
#pragma unroll
    for (int i = 0; i < 8; ++i) z[i] = 0.0f;
    return z;
}

__device__ __forceinline__ v16bf cat8(v8bf lo, v8bf hi) {
    return __builtin_shufflevector(lo, hi, 0,1,2,3,4,5,6,7,8,9,10,11,12,13,14,15);
}

// XOR-swizzled LDS layout for rows x 1024 bf16 tiles: 16-byte chunks (8 bf16)
// chunk' = chunk ^ (row & 15) -> A-fragment ds reads hit all 64 banks.
__device__ __forceinline__ int sw_idx(int row, int col) {
    return row * TWO_D + ((((col >> 3) ^ (row & 15)) & 127) << 3) + (col & 7);
}

// A-fragment (16x32 bf16), lane = m + 16*h, rows [rowbase, rowbase+16).
__device__ __forceinline__ v16bf load_a_sw(const __bf16* buf, int m, int k0, int h) {
    const v8bf* p = (const v8bf*)buf;
    const int c0 = (k0 >> 3) + h;
    v8bf lo = p[m * 128 + ((c0    ) ^ (m & 15))];
    v8bf hi = p[m * 128 + ((c0 + 2) ^ (m & 15))];
    return cat8(lo, hi);
}

// B-fragment (32x16 bf16) from transposed bf16 weights (N x K row-major).
__device__ __forceinline__ v16bf load_b(const __bf16* wt_col, int k0, int h) {
    const v8bf* p = (const v8bf*)(wt_col + k0 + 8 * h);
    return cat8(p[0], p[2]);
}

__device__ __forceinline__ v8f wmma_bf16(v16bf a, v16bf b, v8f c) {
    return __builtin_amdgcn_wmma_f32_16x16x32_bf16(false, a, false, b,
                                                   (short)0, c, false, false);
}

// wave32 argmax with first-max (lowest index) tiebreak, matches jnp.argmax
__device__ __forceinline__ int argmax32(float v, int lane) {
    int idx = lane;
#pragma unroll
    for (int off = 16; off > 0; off >>= 1) {
        float ov = __shfl_xor(v, off, 32);
        int   oi = __shfl_xor(idx, off, 32);
        if (ov > v || (ov == v && oi < idx)) { v = ov; idx = oi; }
    }
    return idx;
}

// -------------------------- kernel 0: weight prep --------------------------
// f32 (K x N) -> bf16 transposed (N x K) in workspace.
__global__ void convert_transpose_kernel(const float* __restrict__ in,
                                         __bf16* __restrict__ out,
                                         int K, int N) {
    int i = blockIdx.x * 256 + threadIdx.x;
    if (i < K * N) {
        int k = i / N, n = i % N;
        out[(size_t)n * K + k] = (__bf16)in[i];
    }
}

// -------------------------- kernel 1: routing ------------------------------
__global__ __launch_bounds__(128) void route_kernel(const float* __restrict__ X,
                                                    const float* __restrict__ Wg,
                                                    const float* __restrict__ Ws,
                                                    int* __restrict__ flat) {
    const int wave = threadIdx.x >> 5;
    const int lane = threadIdx.x & 31;
    const int tok  = blockIdx.x * 4 + wave;
    const float* xr = X + (size_t)tok * D_;
    float ag = 0.0f, as = 0.0f;
    for (int d = 0; d < D_; ++d) {
        float xv = xr[d];
        ag = fmaf(xv, Wg[d * G_ + lane], ag);
        as = fmaf(xv, Ws[d * K_ + lane], as);
    }
    int gi = argmax32(ag, lane);
    int si = argmax32(as, lane);
    if (lane == 0) flat[tok] = gi * K_ + si;
}

// -------------------------- kernel 2: token MLPs + scatter -----------------
__global__ __launch_bounds__(256) void token_kernel(
    const float* __restrict__ X, const float* __restrict__ S,
    const int* __restrict__ src_ids,
    const __bf16* __restrict__ msg_w1t, const float* __restrict__ msg_b1,
    const __bf16* __restrict__ msg_w2t, const float* __restrict__ msg_b2,
    const __bf16* __restrict__ gate_w1t, const float* __restrict__ gate_b1,
    const float* __restrict__ gate_w2, const float* __restrict__ gate_b2,
    const int* __restrict__ flat, float* __restrict__ incoming)
{
    __shared__ __bf16 xsbuf[TM * TWO_D];   // XS tile (swizzled)       64 KB
    __shared__ __bf16 hbuf [TM * TWO_D];   // g1 then h1 (swizzled)    64 KB

    const int tid  = threadIdx.x;
    const int wave = tid >> 5;
    const int lane = tid & 31;
    const int m    = lane & 15;
    const int h    = lane >> 4;

    const int tok0 = blockIdx.x * TM;
    const int b    = tok0 / L_;            // blocks never straddle batches

    // ---- gather-load XS = [X | S[src]] as bf16 into swizzled LDS ----------
    for (int c = tid; c < TM * 128; c += 256) {
        int row = c >> 7, ch = c & 127, col = ch << 3;
        int tok = tok0 + row;
        const float* srcp;
        if (col < D_) {
            srcp = X + (size_t)tok * D_ + col;
        } else {
            int sid = src_ids[tok];
            sid = sid < 0 ? 0 : (sid > N_SLOTS - 1 ? N_SLOTS - 1 : sid);
            srcp = S + ((size_t)b * N_SLOTS + sid) * D_ + (col - D_);
        }
        v8bf v;
#pragma unroll
        for (int i = 0; i < 8; ++i) v[i] = (__bf16)srcp[i];
        ((v8bf*)xsbuf)[row * 128 + (ch ^ (row & 15))] = v;
    }
    __syncthreads();

    // ---- g1 = gelu(XS @ gate_w1 + gate_b1)  (32 x 512), 64 cols per wave --
    {
        v8f acc[2][4];
#pragma unroll
        for (int mt = 0; mt < 2; ++mt)
#pragma unroll
            for (int t = 0; t < 4; ++t) acc[mt][t] = zero8();
        for (int k0 = 0; k0 < TWO_D; k0 += 32) {
            v16bf a0 = load_a_sw(xsbuf,              m, k0, h);
            v16bf a1 = load_a_sw(xsbuf + 16 * TWO_D, m, k0, h);
#pragma unroll
            for (int t = 0; t < 4; ++t) {
                int ncol = wave * 64 + t * 16 + m;
                v16bf bf = load_b(gate_w1t + (size_t)ncol * TWO_D, k0, h);
                acc[0][t] = wmma_bf16(a0, bf, acc[0][t]);
                acc[1][t] = wmma_bf16(a1, bf, acc[1][t]);
            }
        }
#pragma unroll
        for (int mt = 0; mt < 2; ++mt)
#pragma unroll
            for (int t = 0; t < 4; ++t) {
                int ncol = wave * 64 + t * 16 + m;
                float bias = gate_b1[ncol];
#pragma unroll
                for (int r = 0; r < 8; ++r)
                    hbuf[sw_idx(mt * 16 + r + 8 * h, ncol)] =
                        (__bf16)gelu_erf(acc[mt][t][r] + bias);
            }
    }
    __syncthreads();

    // ---- gates: every wave computes all 32 logits, keeps its 16 rows ------
    float gate8[2][8];                     // gate for rows mt*16 + 8h .. +7
    {
        const float gb = gate_b2[0];
#pragma unroll
        for (int t = 0; t < TM; ++t) {
            float s = 0.0f;
#pragma unroll
            for (int j = 0; j < 16; ++j) {
                int col = lane + 32 * j;
                s = fmaf((float)hbuf[sw_idx(t, col)], gate_w2[col], s);
            }
#pragma unroll
            for (int off = 16; off > 0; off >>= 1) s += __shfl_xor(s, off, 32);
            float g = 1.0f / (1.0f + expf(-(s + gb)));
            if (((t & 15) >> 3) == h) gate8[t >> 4][t & 7] = g;
        }
    }
    __syncthreads();                        // g1 fully consumed

    // ---- h1 = gelu(XS @ msg_w1 + msg_b1)  (32 x 1024), 128 cols per wave --
    {
        v8f acc[2][8];
#pragma unroll
        for (int mt = 0; mt < 2; ++mt)
#pragma unroll
            for (int t = 0; t < 8; ++t) acc[mt][t] = zero8();
        for (int k0 = 0; k0 < TWO_D; k0 += 32) {
            v16bf a0 = load_a_sw(xsbuf,              m, k0, h);
            v16bf a1 = load_a_sw(xsbuf + 16 * TWO_D, m, k0, h);
#pragma unroll
            for (int t = 0; t < 8; ++t) {
                int ncol = wave * 128 + t * 16 + m;
                v16bf bf = load_b(msg_w1t + (size_t)ncol * TWO_D, k0, h);
                acc[0][t] = wmma_bf16(a0, bf, acc[0][t]);
                acc[1][t] = wmma_bf16(a1, bf, acc[1][t]);
            }
        }
#pragma unroll
        for (int mt = 0; mt < 2; ++mt)
#pragma unroll
            for (int t = 0; t < 8; ++t) {
                int ncol = wave * 128 + t * 16 + m;
                float bias = msg_b1[ncol];
#pragma unroll
                for (int r = 0; r < 8; ++r)
                    hbuf[sw_idx(mt * 16 + r + 8 * h, ncol)] =
                        (__bf16)gelu_erf(acc[mt][t][r] + bias);
            }
    }
    __syncthreads();

    // ---- msg = h1 @ msg_w2 + msg_b2 ; gated scatter-add -------------------
    {
        v8f acc[2][4];
#pragma unroll
        for (int mt = 0; mt < 2; ++mt)
#pragma unroll
            for (int t = 0; t < 4; ++t) acc[mt][t] = zero8();
        for (int k0 = 0; k0 < TWO_D; k0 += 32) {
            v16bf a0 = load_a_sw(hbuf,              m, k0, h);
            v16bf a1 = load_a_sw(hbuf + 16 * TWO_D, m, k0, h);
#pragma unroll
            for (int t = 0; t < 4; ++t) {
                int ncol = wave * 64 + t * 16 + m;
                v16bf bf = load_b(msg_w2t + (size_t)ncol * TWO_D, k0, h);
                acc[0][t] = wmma_bf16(a0, bf, acc[0][t]);
                acc[1][t] = wmma_bf16(a1, bf, acc[1][t]);
            }
        }
        int fl[2][8];
#pragma unroll
        for (int mt = 0; mt < 2; ++mt)
#pragma unroll
            for (int r = 0; r < 8; ++r)
                fl[mt][r] = flat[tok0 + mt * 16 + r + 8 * h];
#pragma unroll
        for (int mt = 0; mt < 2; ++mt)
#pragma unroll
            for (int t = 0; t < 4; ++t) {
                int ncol = wave * 64 + t * 16 + m;
                float bias = msg_b2[ncol];
#pragma unroll
                for (int r = 0; r < 8; ++r) {
                    float v = (acc[mt][t][r] + bias) * gate8[mt][r];
                    atomicAdd(&incoming[((size_t)b * N_SLOTS + fl[mt][r]) * D_ + ncol], v);
                }
            }
    }
}

// -------------------------- kernel 3: slot update + LN ---------------------
__global__ __launch_bounds__(256) void slot_kernel(
    const float* __restrict__ S, const float* __restrict__ incoming,
    const __bf16* __restrict__ upd_w1t, const float* __restrict__ upd_b1,
    const __bf16* __restrict__ upd_w2t, const float* __restrict__ upd_b2,
    const float* __restrict__ ln_g, const float* __restrict__ ln_b,
    float* __restrict__ out)
{
    __shared__ __bf16 sibuf[TM * TWO_D];   // SI tile, reused as f32 S_new
    __shared__ __bf16 hbuf [TM * TWO_D];

    const int tid  = threadIdx.x;
    const int wave = tid >> 5;
    const int lane = tid & 31;
    const int m    = lane & 15;
    const int h    = lane >> 4;
    const int row0 = blockIdx.x * TM;      // rows of flattened (B*N, D)

    // ---- load SI = [S | incoming] as bf16, swizzled -----------------------
    for (int c = tid; c < TM * 128; c += 256) {
        int r = c >> 7, ch = c & 127, col = ch << 3;
        size_t base = (size_t)(row0 + r) * D_;
        const float* srcp = (col < D_) ? (S + base + col)
                                       : (incoming + base + (col - D_));
        v8bf v;
#pragma unroll
        for (int i = 0; i < 8; ++i) v[i] = (__bf16)srcp[i];
        ((v8bf*)sibuf)[r * 128 + (ch ^ (r & 15))] = v;
    }
    __syncthreads();

    // ---- hidden = gelu(SI @ upd_w1 + upd_b1)  (32 x 1024) -----------------
    {
        v8f acc[2][8];
#pragma unroll
        for (int mt = 0; mt < 2; ++mt)
#pragma unroll
            for (int t = 0; t < 8; ++t) acc[mt][t] = zero8();
        for (int k0 = 0; k0 < TWO_D; k0 += 32) {
            v16bf a0 = load_a_sw(sibuf,              m, k0, h);
            v16bf a1 = load_a_sw(sibuf + 16 * TWO_D, m, k0, h);
#pragma unroll
            for (int t = 0; t < 8; ++t) {
                int ncol = wave * 128 + t * 16 + m;
                v16bf bf = load_b(upd_w1t + (size_t)ncol * TWO_D, k0, h);
                acc[0][t] = wmma_bf16(a0, bf, acc[0][t]);
                acc[1][t] = wmma_bf16(a1, bf, acc[1][t]);
            }
        }
#pragma unroll
        for (int mt = 0; mt < 2; ++mt)
#pragma unroll
            for (int t = 0; t < 8; ++t) {
                int ncol = wave * 128 + t * 16 + m;
                float bias = upd_b1[ncol];
#pragma unroll
                for (int r = 0; r < 8; ++r)
                    hbuf[sw_idx(mt * 16 + r + 8 * h, ncol)] =
                        (__bf16)gelu_erf(acc[mt][t][r] + bias);
            }
    }
    __syncthreads();

    // ---- S_new = S + hidden @ upd_w2 + upd_b2  -> sibuf (f32, 32 x 512) ---
    float* snew = (float*)sibuf;
    {
        v8f acc[2][4];
#pragma unroll
        for (int mt = 0; mt < 2; ++mt)
#pragma unroll
            for (int t = 0; t < 4; ++t) acc[mt][t] = zero8();
        for (int k0 = 0; k0 < TWO_D; k0 += 32) {
            v16bf a0 = load_a_sw(hbuf,              m, k0, h);
            v16bf a1 = load_a_sw(hbuf + 16 * TWO_D, m, k0, h);
#pragma unroll
            for (int t = 0; t < 4; ++t) {
                int ncol = wave * 64 + t * 16 + m;
                v16bf bf = load_b(upd_w2t + (size_t)ncol * TWO_D, k0, h);
                acc[0][t] = wmma_bf16(a0, bf, acc[0][t]);
                acc[1][t] = wmma_bf16(a1, bf, acc[1][t]);
            }
        }
#pragma unroll
        for (int mt = 0; mt < 2; ++mt)
#pragma unroll
            for (int t = 0; t < 4; ++t) {
                int ncol = wave * 64 + t * 16 + m;
                float bias = upd_b2[ncol];
#pragma unroll
                for (int r = 0; r < 8; ++r) {
                    int row = mt * 16 + r + 8 * h;
                    float v = S[(size_t)(row0 + row) * D_ + ncol] + acc[mt][t][r] + bias;
                    snew[row * D_ + ncol] = v;
                }
            }
    }
    __syncthreads();

    // ---- LayerNorm over D per row: 4 rows per wave ------------------------
#pragma unroll
    for (int rr = 0; rr < 4; ++rr) {
        int row = wave * 4 + rr;
        float s = 0.0f, s2 = 0.0f;
        float vals[16];
#pragma unroll
        for (int j = 0; j < 16; ++j) {
            float v = snew[row * D_ + lane + 32 * j];
            vals[j] = v;
            s += v;
            s2 = fmaf(v, v, s2);
        }
#pragma unroll
        for (int off = 16; off > 0; off >>= 1) {
            s  += __shfl_xor(s,  off, 32);
            s2 += __shfl_xor(s2, off, 32);
        }
        const float mu  = s * (1.0f / (float)D_);
        const float var = s2 * (1.0f / (float)D_) - mu * mu;
        const float inv = rsqrtf(var + LN_EPS);
        size_t obase = (size_t)(row0 + row) * D_;
#pragma unroll
        for (int j = 0; j < 16; ++j) {
            int col = lane + 32 * j;
            out[obase + col] = (vals[j] - mu) * inv * ln_g[col] + ln_b[col];
        }
    }
}

// -------------------------- host launcher ----------------------------------

extern "C" void kernel_launch(void* const* d_in, const int* in_sizes, int n_in,
                              void* d_out, int out_size, void* d_ws, size_t ws_size,
                              hipStream_t stream)
{
    (void)in_sizes; (void)n_in; (void)out_size; (void)ws_size;

    const float* X       = (const float*)d_in[0];
    const float* S       = (const float*)d_in[1];
    const int*   src_ids = (const int*)  d_in[2];
    const float* Wg      = (const float*)d_in[3];
    const float* Ws      = (const float*)d_in[4];
    const float* msg_w1  = (const float*)d_in[5];
    const float* msg_b1  = (const float*)d_in[6];
    const float* msg_w2  = (const float*)d_in[7];
    const float* msg_b2  = (const float*)d_in[8];
    const float* gate_w1 = (const float*)d_in[9];
    const float* gate_b1 = (const float*)d_in[10];
    const float* gate_w2 = (const float*)d_in[11];
    const float* gate_b2 = (const float*)d_in[12];
    const float* upd_w1  = (const float*)d_in[13];
    const float* upd_b1  = (const float*)d_in[14];
    const float* upd_w2  = (const float*)d_in[15];
    const float* upd_b2  = (const float*)d_in[16];
    const float* ln_g    = (const float*)d_in[17];
    const float* ln_b    = (const float*)d_in[18];

    char* ws = (char*)d_ws;
    __bf16* msg_w1t  = (__bf16*)ws; ws += (size_t)TWO_D * TWO_D * 2;
    __bf16* msg_w2t  = (__bf16*)ws; ws += (size_t)D_    * TWO_D * 2;
    __bf16* gate_w1t = (__bf16*)ws; ws += (size_t)D_    * TWO_D * 2;
    __bf16* upd_w1t  = (__bf16*)ws; ws += (size_t)TWO_D * TWO_D * 2;
    __bf16* upd_w2t  = (__bf16*)ws; ws += (size_t)D_    * TWO_D * 2;
    float*  incoming = (float*) ws; ws += (size_t)B_ * N_SLOTS * D_ * 4;
    int*    flat     = (int*)   ws; ws += (size_t)B_ * L_ * 4;

    // 0) bf16 transposed weight copies (fit in 192MB L2; redone every call)
    {
        int t;
        t = TWO_D * TWO_D;
        convert_transpose_kernel<<<(t + 255) / 256, 256, 0, stream>>>(msg_w1, msg_w1t, TWO_D, TWO_D);
        t = TWO_D * D_;
        convert_transpose_kernel<<<(t + 255) / 256, 256, 0, stream>>>(msg_w2, msg_w2t, TWO_D, D_);
        convert_transpose_kernel<<<(t + 255) / 256, 256, 0, stream>>>(gate_w1, gate_w1t, TWO_D, D_);
        t = TWO_D * TWO_D;
        convert_transpose_kernel<<<(t + 255) / 256, 256, 0, stream>>>(upd_w1, upd_w1t, TWO_D, TWO_D);
        t = TWO_D * D_;
        convert_transpose_kernel<<<(t + 255) / 256, 256, 0, stream>>>(upd_w2, upd_w2t, TWO_D, D_);
    }

    // 1) zero the scatter-add accumulator
    hipMemsetAsync(incoming, 0, (size_t)B_ * N_SLOTS * D_ * sizeof(float), stream);

    // 2) hard hierarchical routing (argmax over 32-wide projections)
    route_kernel<<<B_ * L_ / 4, 128, 0, stream>>>(X, Wg, Ws, flat);

    // 3) token MLPs (msg + gate) + gated atomic scatter into slots
    token_kernel<<<B_ * L_ / TM, 256, 0, stream>>>(
        X, S, src_ids,
        msg_w1t, msg_b1, msg_w2t, msg_b2,
        gate_w1t, gate_b1, gate_w2, gate_b2,
        flat, incoming);

    // 4) slot update MLP + residual + LayerNorm
    slot_kernel<<<B_ * N_SLOTS / TM, 256, 0, stream>>>(
        S, incoming, upd_w1t, upd_b1, upd_w2t, upd_b2, ln_g, ln_b,
        (float*)d_out);
}
